// InvariantEdgeConv_60189671686871
// MI455X (gfx1250) — matrix operand order repaired and checked by reading, not compiled, not run on hardware
//
#include <hip/hip_runtime.h>
#include <math.h>

typedef __attribute__((ext_vector_type(16))) __bf16 v16bf;
typedef __attribute__((ext_vector_type(8)))  float  v8f;
typedef unsigned short ushort_t;

#define B_  4
#define N_  16384
#define K_  16
#define C_  64
#define H_  128
#define O_  128
#define E_  133
#define EP_ 160            // E padded to multiple of 32
#define NPTS (B_ * N_)     // 65536 points
#define NS   (B_ * N_ * K_)  // 1,048,576 BN samples
#define EPS_   1e-6f
#define BNEPS_ 1e-5f
#define LNEPS_ 1e-5f

// ---- workspace layout (bytes) ----
#define WS_STATS   0            // 512 f32: [sum1(128) sumsq1(128) sum2(128) sumsq2(128)]
#define WS_PARAMS  2048         // 512 f32: [sc1 sh1 sc2 sh2]
#define WS_W1BF    4096         // 128*160 bf16 (zero padded cols)
#define WS_W2BF    45056        // 128*128 bf16
#define WS_POOLED  77824        // NPTS*128 f32 raw pooled (pre-BN2)

// ---- LDS partition (bytes) for fused kernel ----
#define L_A1    0               // 256*160 bf16 = 81920
#define L_W1    81920           // 128*160 bf16 = 40960
#define L_STATS 122880          // 256 f32     = 1024
#define L_A2    123904          // 256*128 bf16 = 65536
#define L_W2    189440          // 128*128 bf16 = 32768
#define LDS_SZ0 123904
#define LDS_SZ1 222208

__device__ __forceinline__ ushort_t f2bf(float f) {
  unsigned u = __float_as_uint(f);
  unsigned r = u + 0x7FFFu + ((u >> 16) & 1u);   // round-to-nearest-even
  return (ushort_t)(r >> 16);
}

union ABFrag { uint4 q[2]; v16bf v; };

// A fragment 16x32 bf16 from row-major LDS [row][k], stride ldk (elements).
// ISA layout: lanes 0-15 row M=lane, K chunks [k0+8h,+8) and [k0+16+8h,+8), h=lane>>4.
__device__ __forceinline__ v16bf load_a_frag(const ushort_t* lds, int row0, int k0,
                                             int ldk, int lane) {
  int r = lane & 15, h = lane >> 4;
  const ushort_t* p = lds + (row0 + r) * ldk + k0 + 8 * h;
  ABFrag u;
  u.q[0] = *(const uint4*)p;
  u.q[1] = *(const uint4*)(p + 16);
  return u.v;
}

// B fragment 32x16 bf16: lane holds column N=lane&15, K chunks [k0+8h,+8) and [k0+16+8h,+8).
// Weights stored row-major [out_channel][k] with stride ldk.
__device__ __forceinline__ v16bf load_b_frag(const ushort_t* lds, int col0, int k0,
                                             int ldk, int lane) {
  int c = lane & 15, h = lane >> 4;
  const ushort_t* p = lds + (col0 + c) * ldk + k0 + 8 * h;
  ABFrag u;
  u.q[0] = *(const uint4*)p;
  u.q[1] = *(const uint4*)(p + 16);
  return u.v;
}

// ---------------- prep: zero stats, convert weights to bf16 ----------------
__global__ void prep_kernel(const float* __restrict__ W1, const float* __restrict__ W2,
                            float* __restrict__ ws_f32, ushort_t* __restrict__ w1bf,
                            ushort_t* __restrict__ w2bf) {
  int tid = threadIdx.x;
  for (int i = tid; i < 512; i += 256) ws_f32[i] = 0.f;           // stats
  for (int i = tid; i < H_ * EP_; i += 256) {
    int h = i / EP_, e = i % EP_;
    w1bf[i] = (e < E_) ? f2bf(W1[h * E_ + e]) : (ushort_t)0;
  }
  for (int i = tid; i < O_ * H_; i += 256) w2bf[i] = f2bf(W2[i]);
}

// ---------------- BN params: stats -> scale/shift ----------------
__global__ void bn_params_kernel(const float* __restrict__ sums,  // [sum(128) sumsq(128)]
                                 const float* __restrict__ g, const float* __restrict__ b,
                                 float* __restrict__ scale, float* __restrict__ shift,
                                 float inv_n, float eps) {
  int c = threadIdx.x;
  float mean = sums[c] * inv_n;
  float var  = sums[128 + c] * inv_n - mean * mean;
  float s = rsqrtf(var + eps) * g[c];
  scale[c] = s;
  shift[c] = b[c] - mean * s;
}

// ---------------- fused GEMM kernel ----------------
// MODE 0: edge-build + GEMM1 -> accumulate BN1 stats.
// MODE 1: edge-build + GEMM1 -> BN1+SiLU -> GEMM2 -> BN2 stats + K-pool.
template <int MODE>
__global__ __launch_bounds__(256) void fused_kernel(
    const float* __restrict__ feat, const float* __restrict__ xyz,
    const int* __restrict__ idx_knn, const ushort_t* __restrict__ w1bf,
    const ushort_t* __restrict__ w2bf, const float* __restrict__ sc1,
    const float* __restrict__ sh1, float* __restrict__ gstats,
    float* __restrict__ pooled) {
  extern __shared__ char smem[];
  ushort_t* sA1 = (ushort_t*)(smem + L_A1);
  ushort_t* sW1 = (ushort_t*)(smem + L_W1);
  float*    sSt = (float*)(smem + L_STATS);
  ushort_t* sA2 = (ushort_t*)(smem + L_A2);
  ushort_t* sW2 = (ushort_t*)(smem + L_W2);

  const int tid  = threadIdx.x;
  const int lane = tid & 31;
  const int wave = tid >> 5;
  const int wm   = wave >> 1;  // 0..3 : 64 rows each
  const int wn   = wave & 1;   // 0..1 : 64 cols each

  // ---- stage weights into LDS + zero LDS stats ----
  for (int i = tid; i < H_ * EP_; i += 256) sW1[i] = w1bf[i];
  if (MODE == 1)
    for (int i = tid; i < O_ * H_; i += 256) sW2[i] = w2bf[i];
  sSt[tid] = 0.f;

  // ---- build one edge row (160 bf16) per thread ----
  {
    const int e  = tid;                 // 0..255
    const int p  = e >> 4;              // local point
    const int kk = e & 15;              // neighbor slot
    const int gp = blockIdx.x * 16 + p; // global point
    const int b  = gp >> 14;            // /N_
    const int n  = gp & (N_ - 1);
    const int jn = idx_knn[((size_t)b * N_ + n) * K_ + kk];

    const float4* c4 = (const float4*)(feat + ((size_t)b * N_ + n) * C_);
    const float4* n4 = (const float4*)(feat + ((size_t)b * N_ + jn) * C_);
    ushort_t* row = sA1 + e * EP_;
#pragma unroll
    for (int q = 0; q < 16; ++q) {
      float4 cv = c4[q], nv = n4[q];
      row[q * 4 + 0] = f2bf(cv.x); row[q * 4 + 1] = f2bf(cv.y);
      row[q * 4 + 2] = f2bf(cv.z); row[q * 4 + 3] = f2bf(cv.w);
      row[64 + q * 4 + 0] = f2bf(nv.x - cv.x); row[64 + q * 4 + 1] = f2bf(nv.y - cv.y);
      row[64 + q * 4 + 2] = f2bf(nv.z - cv.z); row[64 + q * 4 + 3] = f2bf(nv.w - cv.w);
    }
    const float* cx = xyz + ((size_t)b * N_ + n) * 3;
    const float* nx = xyz + ((size_t)b * N_ + jn) * 3;
    float c0 = cx[0], c1 = cx[1], c2 = cx[2];
    float x0 = nx[0], x1 = nx[1], x2 = nx[2];
    float rc  = sqrtf(c0 * c0 + c1 * c1 + c2 * c2);
    float rn  = sqrtf(x0 * x0 + x1 * x1 + x2 * x2);
    float d0 = x0 - c0, d1 = x1 - c1, d2 = x2 - c2;
    float rr  = sqrtf(d0 * d0 + d1 * d1 + d2 * d2);
    float dot = c0 * x0 + c1 * x1 + c2 * x2;
    float cosv = dot / (rc * rn + EPS_);
    row[128] = f2bf(rc); row[129] = f2bf(rn); row[130] = f2bf(rr);
    row[131] = f2bf(dot); row[132] = f2bf(cosv);
#pragma unroll
    for (int t = E_; t < EP_; ++t) row[t] = 0;
  }
  __syncthreads();

  // ---- GEMM1: (256 x 160) @ (160 x 128), wave computes 64x64 ----
  v8f acc[4][4];
#pragma unroll
  for (int i = 0; i < 4; ++i)
#pragma unroll
    for (int j = 0; j < 4; ++j)
#pragma unroll
      for (int t = 0; t < 8; ++t) acc[i][j][t] = 0.f;

#pragma unroll
  for (int ks = 0; ks < EP_; ks += 32) {
    v16bf af[4], bf[4];
#pragma unroll
    for (int i = 0; i < 4; ++i) af[i] = load_a_frag(sA1, wm * 64 + i * 16, ks, EP_, lane);
#pragma unroll
    for (int j = 0; j < 4; ++j) bf[j] = load_b_frag(sW1, wn * 64 + j * 16, ks, EP_, lane);
#pragma unroll
    for (int i = 0; i < 4; ++i)
#pragma unroll
      for (int j = 0; j < 4; ++j)
        acc[i][j] = __builtin_amdgcn_wmma_f32_16x16x32_bf16(
            false, af[i], false, bf[j], (short)0, acc[i][j], false, false);
  }

  if (MODE == 0) {
    // ---- BN1 stats: per-channel sum / sumsq over this block's 256 rows ----
#pragma unroll
    for (int j = 0; j < 4; ++j) {
      float s = 0.f, s2 = 0.f;
#pragma unroll
      for (int i = 0; i < 4; ++i)
#pragma unroll
        for (int v = 0; v < 8; ++v) {
          float x = acc[i][j][v];
          s += x; s2 += x * x;
        }
      s  += __shfl_xor(s, 16, 32);   // join M halves (lane ^ 16 = same column)
      s2 += __shfl_xor(s2, 16, 32);
      if ((lane >> 4) == 0) {
        int ch = wn * 64 + j * 16 + lane;
        atomicAdd(&sSt[ch], s);          // ds_add_f32
        atomicAdd(&sSt[128 + ch], s2);
      }
    }
    __syncthreads();
    atomicAdd(&gstats[tid], sSt[tid]);   // one global atomic per channel-stat
    return;
  }

  // ---- MODE 1: BN1 + SiLU, stage bf16 activations for GEMM2 ----
  {
    float s1v[4], s1h[4];
#pragma unroll
    for (int j = 0; j < 4; ++j) {
      int ch = wn * 64 + j * 16 + (lane & 15);
      s1v[j] = sc1[ch]; s1h[j] = sh1[ch];
    }
    const int h = lane >> 4;
#pragma unroll
    for (int i = 0; i < 4; ++i)
#pragma unroll
      for (int j = 0; j < 4; ++j) {
        int ch = wn * 64 + j * 16 + (lane & 15);
#pragma unroll
        for (int v = 0; v < 8; ++v) {
          float x = acc[i][j][v] * s1v[j] + s1h[j];
          float y = x / (1.f + __expf(-x));  // SiLU
          int row = wm * 64 + i * 16 + v + 8 * h;
          sA2[row * H_ + ch] = f2bf(y);
        }
      }
  }
  __syncthreads();

  // ---- GEMM2: (256 x 128) @ (128 x 128) ----
  v8f acc2[4][4];
#pragma unroll
  for (int i = 0; i < 4; ++i)
#pragma unroll
    for (int j = 0; j < 4; ++j)
#pragma unroll
      for (int t = 0; t < 8; ++t) acc2[i][j][t] = 0.f;

#pragma unroll
  for (int ks = 0; ks < H_; ks += 32) {
    v16bf af[4], bf[4];
#pragma unroll
    for (int i = 0; i < 4; ++i) af[i] = load_a_frag(sA2, wm * 64 + i * 16, ks, H_, lane);
#pragma unroll
    for (int j = 0; j < 4; ++j) bf[j] = load_b_frag(sW2, wn * 64 + j * 16, ks, H_, lane);
#pragma unroll
    for (int i = 0; i < 4; ++i)
#pragma unroll
      for (int j = 0; j < 4; ++j)
        acc2[i][j] = __builtin_amdgcn_wmma_f32_16x16x32_bf16(
            false, af[i], false, bf[j], (short)0, acc2[i][j], false, false);
  }

  // ---- BN2 stats + K-pool (one 16-row tile == one point's 16 edges) ----
#pragma unroll
  for (int j = 0; j < 4; ++j) {
    float cs = 0.f, cs2 = 0.f;
#pragma unroll
    for (int i = 0; i < 4; ++i) {
      float s = 0.f, s2 = 0.f;
#pragma unroll
      for (int v = 0; v < 8; ++v) {
        float x = acc2[i][j][v];
        s += x; s2 += x * x;
      }
      s  += __shfl_xor(s, 16, 32);
      s2 += __shfl_xor(s2, 16, 32);
      cs += s; cs2 += s2;
      if ((lane >> 4) == 0) {
        int gp = blockIdx.x * 16 + wm * 4 + i;
        int ch = wn * 64 + j * 16 + lane;
        pooled[(size_t)gp * O_ + ch] = s * (1.f / 16.f);  // raw mean over K
      }
    }
    if ((lane >> 4) == 0) {
      int ch = wn * 64 + j * 16 + lane;
      atomicAdd(&sSt[ch], cs);
      atomicAdd(&sSt[128 + ch], cs2);
    }
  }
  __syncthreads();
  atomicAdd(&gstats[256 + tid], sSt[tid]);
}

// ---------------- finalize: shortcut GEMM + LN + BN2-affine + SiLU ----------------
__global__ __launch_bounds__(128) void finalize_kernel(
    const float* __restrict__ feat, const float* __restrict__ Wsc,
    const float* __restrict__ ln_g, const float* __restrict__ ln_b,
    const float* __restrict__ sc2, const float* __restrict__ sh2,
    const float* __restrict__ pooled, float* __restrict__ out) {
  __shared__ float sWsc[O_ * C_];   // 32 KB
  __shared__ float sFeat[16 * C_];  //  4 KB
  const int tid = threadIdx.x, lane = tid & 31, wave = tid >> 5;

  for (int i = tid; i < O_ * C_; i += 128) sWsc[i] = Wsc[i];
  for (int i = tid; i < 16 * C_; i += 128) sFeat[i] = feat[(size_t)blockIdx.x * 16 * C_ + i];
  __syncthreads();

#pragma unroll
  for (int pp = 0; pp < 4; ++pp) {          // each wave handles 4 points
    const int lp = wave * 4 + pp;
    const int pt = blockIdx.x * 16 + lp;
    float v[4] = {0.f, 0.f, 0.f, 0.f};      // channels lane + 32*q
    for (int c = 0; c < C_; ++c) {
      float f = sFeat[lp * C_ + c];
#pragma unroll
      for (int q = 0; q < 4; ++q) v[q] += sWsc[(lane + 32 * q) * C_ + c] * f;
    }
    float s1 = 0.f, s2 = 0.f;
#pragma unroll
    for (int q = 0; q < 4; ++q) { s1 += v[q]; s2 += v[q] * v[q]; }
#pragma unroll
    for (int m = 16; m >= 1; m >>= 1) {
      s1 += __shfl_xor(s1, m, 32);
      s2 += __shfl_xor(s2, m, 32);
    }
    float mu = s1 * (1.f / 128.f);
    float var = s2 * (1.f / 128.f) - mu * mu;
    float rs = rsqrtf(var + LNEPS_);
#pragma unroll
    for (int q = 0; q < 4; ++q) {
      int ch = lane + 32 * q;
      float scv = (v[q] - mu) * rs * ln_g[ch] + ln_b[ch];
      float pb = pooled[(size_t)pt * O_ + ch] * sc2[ch] + sh2[ch];
      float z = pb + scv;
      out[(size_t)pt * O_ + ch] = z / (1.f + __expf(-z));
    }
  }
}

extern "C" void kernel_launch(void* const* d_in, const int* in_sizes, int n_in,
                              void* d_out, int out_size, void* d_ws, size_t ws_size,
                              hipStream_t stream) {
  const float* feat = (const float*)d_in[0];
  const float* xyz  = (const float*)d_in[1];
  const float* W1   = (const float*)d_in[2];
  const float* g1   = (const float*)d_in[3];
  const float* b1   = (const float*)d_in[4];
  const float* W2   = (const float*)d_in[5];
  const float* g2   = (const float*)d_in[6];
  const float* b2   = (const float*)d_in[7];
  const float* Wsc  = (const float*)d_in[8];
  const float* lng  = (const float*)d_in[9];
  const float* lnb  = (const float*)d_in[10];
  const int*   idx  = (const int*)d_in[11];
  float* out = (float*)d_out;

  char* ws = (char*)d_ws;
  float*    gstats = (float*)(ws + WS_STATS);
  float*    params = (float*)(ws + WS_PARAMS);
  float*    sc1 = params, *sh1 = params + 128, *sc2 = params + 256, *sh2 = params + 384;
  ushort_t* w1bf = (ushort_t*)(ws + WS_W1BF);
  ushort_t* w2bf = (ushort_t*)(ws + WS_W2BF);
  float*    pooled = (float*)(ws + WS_POOLED);

  const int nblk = NPTS / 16;  // 4096
  const float inv_ns = 1.f / (float)NS;

  prep_kernel<<<1, 256, 0, stream>>>(W1, W2, gstats, w1bf, w2bf);

  fused_kernel<0><<<nblk, 256, LDS_SZ0, stream>>>(feat, xyz, idx, w1bf, w2bf,
                                                  sc1, sh1, gstats, pooled);
  bn_params_kernel<<<1, 128, 0, stream>>>(gstats, g1, b1, sc1, sh1, inv_ns, BNEPS_);

  fused_kernel<1><<<nblk, 256, LDS_SZ1, stream>>>(feat, xyz, idx, w1bf, w2bf,
                                                  sc1, sh1, gstats, pooled);
  bn_params_kernel<<<1, 128, 0, stream>>>(gstats + 256, g2, b2, sc2, sh2, inv_ns, BNEPS_);

  finalize_kernel<<<nblk, 128, 0, stream>>>(feat, Wsc, lng, lnb, sc2, sh2, pooled, out);
}
